// SimpleBiLSTM_18751827215079
// MI455X (gfx1250) — matrix operand order repaired and checked by reading, not compiled, not run on hardware
//
#include <hip/hip_runtime.h>
#include <hip/hip_bf16.h>
#include <stdint.h>

// Bidirectional LSTM for MI455X (gfx1250, wave32, WMMA).
// B=32, T=4096, F=64, H=256, OUT=64.
//
// Kernel 1: prepack  — bf16-pack [Wi;Wh] (K=320 x N=1024) and Wd halves
//                      into WMMA B-fragment layout in d_ws (L2-resident).
// Kernel 2: bilstm   — 2 blocks (fwd/bwd), 1024 threads = 32 waves each.
//                      Per step: stage x_t (bf16) -> 1280 v_wmma bf16 ops
//                      for the gate GEMM -> fused activations -> fused
//                      output projection (WMMA) scattered to out / partial.
//                      Weight *offsets* are laundered per timestep: the
//                      addresses stay provenance-linked to the global
//                      kernarg pointer (global_load_b128, LOADcnt only)
//                      but become loop-variant, so the fragments stream
//                      from L2 instead of being hoisted into 160 VGPRs
//                      and spilled to scratch.
// Kernel 3: add_bwd  — out += backward partial (deterministic, no atomics).

#define B_   32
#define T_   4096
#define F_   64
#define H_   256
#define G_   1024   // 4*H
#define K_   320    // F + H
#define OUT_ 64
#define KC_  10     // K chunks of 32 for gate GEMM
#define NT_  64     // N tiles of 16 for gate GEMM
#define DKC_ 8      // dense K chunks (256/32)
#define DNT_ 4      // dense N tiles (64/16)

#define W_ELEMS_PER_DIR  (NT_ * KC_ * 32 * 16)          // 327680 bf16
#define WD_ELEMS_PER_DIR (DNT_ * DKC_ * 32 * 16)        // 16384 bf16
#define WD_BASE_HALF     (2 * W_ELEMS_PER_DIR)          // 655360
#define PB_BYTE_OFF      ((2 * W_ELEMS_PER_DIR + 2 * WD_ELEMS_PER_DIR) * 2)  // 1376256
#define OUTS_OFF         (4 * B_ * H_)                  // 32768 floats (4 carries first)

typedef __attribute__((ext_vector_type(16))) __bf16 v16bf;
typedef __attribute__((ext_vector_type(8)))  __bf16 v8bf;
typedef __attribute__((ext_vector_type(8)))  float  v8f;

// ---- WMMA helpers -----------------------------------------------------------

static __device__ inline v8f wmma_bf16(v16bf a, v16bf b, v8f c) {
  // D = A(16x32 bf16) * B(32x16 bf16) + C(16x16 f32)
  return __builtin_amdgcn_wmma_f32_16x16x32_bf16(
      /*neg_a=*/false, a, /*neg_b=*/false, b,
      /*c_mod=*/(short)0, c, /*reuse_a=*/false, /*reuse_b=*/false);
}

// A-fragment (16x32 bf16): lane L holds row m=L%16;
// lanes 0-15: K = {0..15}; lanes 16-31: K = {16..31}.
// Both K runs are contiguous -> two 16B LDS loads per fragment.
static __device__ inline v16bf load_a_frag(const __bf16* a_row, int kbase, int lane) {
  const int kb = kbase + ((lane & 16) ? 16 : 0);
  v8bf lo = *(const v8bf*)(a_row + kb);
  v8bf hi = *(const v8bf*)(a_row + kb + 8);
  union { v16bf v; v8bf h[2]; } u;
  u.h[0] = lo; u.h[1] = hi;
  return u.v;
}

// B-fragment (32x16 bf16): packed so each lane's 16 halves are contiguous
// (two global_load_b128 per fragment). Lane L: col n=L%16, K=(L>=16?16:0)+e.
static __device__ inline v16bf load_b_frag(const __bf16* p) {
  v8bf lo = *(const v8bf*)(p);
  v8bf hi = *(const v8bf*)(p + 8);
  union { v16bf v; v8bf h[2]; } u;
  u.h[0] = lo; u.h[1] = hi;
  return u.v;
}

// Zero-cost optimization barrier on an *offset* (not a pointer): the address
// wsb + launder_off(...) keeps global-pointer provenance (lowers to
// global_load_b128), but is loop-variant each timestep, so the weight
// fragments cannot be hoisted out of the recurrence and spilled.
static __device__ inline size_t launder_off(size_t v) {
  asm volatile("" : "+s"(v));
  return v;
}

static __device__ inline float sigf(float x) {
  return 1.0f / (1.0f + __expf(-x));
}
static __device__ inline float tanhfast(float x) {
  return 2.0f / (1.0f + __expf(-2.0f * x)) - 1.0f;
}

// ---- Kernel 1: weight pre-pack (f32 -> bf16 WMMA B-fragment layout) ---------

__global__ void lstm_prepack(const float* __restrict__ Wi_f, const float* __restrict__ Wh_f,
                             const float* __restrict__ Wi_b, const float* __restrict__ Wh_b,
                             const float* __restrict__ Wd, __bf16* __restrict__ wsb) {
  const int idx = blockIdx.x * blockDim.x + threadIdx.x;
  const int WTOT = 2 * W_ELEMS_PER_DIR;
  if (idx < WTOT) {
    const int dir  = idx / W_ELEMS_PER_DIR;
    const int r    = idx % W_ELEMS_PER_DIR;
    const int frag = r >> 9;             // (ntile*KC_ + kc)
    const int lane = (r >> 4) & 31;
    const int e    = r & 15;
    const int ntile = frag / KC_;
    const int kc    = frag % KC_;
    const int k = kc * 32 + ((lane & 16) ? 16 : 0) + e;   // 0..319
    const int n = ntile * 16 + (lane & 15);               // 0..1023
    const float* Wi = dir ? Wi_b : Wi_f;
    const float* Wh = dir ? Wh_b : Wh_f;
    const float v = (k < F_) ? Wi[k * G_ + n] : Wh[(k - F_) * G_ + n];
    wsb[idx] = (__bf16)v;
  } else if (idx < WTOT + 2 * WD_ELEMS_PER_DIR) {
    const int j    = idx - WTOT;
    const int dir  = j / WD_ELEMS_PER_DIR;
    const int r    = j % WD_ELEMS_PER_DIR;
    const int frag = r >> 9;             // (ntile*DKC_ + kc)
    const int lane = (r >> 4) & 31;
    const int e    = r & 15;
    const int ntile = frag / DKC_;
    const int kc    = frag % DKC_;
    const int k = dir * H_ + kc * 32 + ((lane & 16) ? 16 : 0) + e;  // row of Wd[512][64]
    const int n = ntile * 16 + (lane & 15);
    wsb[idx] = (__bf16)Wd[k * OUT_ + n];
  }
}

// ---- Kernel 2: persistent recurrent kernel, one block per direction ---------

__global__ __launch_bounds__(1024, 1)
void bilstm_recurrent(const float* __restrict__ x, const int* __restrict__ lens,
                      const float* __restrict__ c0f, const float* __restrict__ h0f,
                      const float* __restrict__ c0b, const float* __restrict__ h0b,
                      const float* __restrict__ bias_f, const float* __restrict__ bias_b,
                      const float* __restrict__ bd,
                      const __bf16* __restrict__ wsb,
                      float* __restrict__ Pb, float* __restrict__ out) {
  __shared__ __bf16 a_buf[B_][K_];     // [x_t(64) | h(256)] bf16 : 20 KB
  __shared__ float  gates[B_][G_];     // 128 KB
  __shared__ float  cbuf[B_][H_];      // 32 KB
  __shared__ float  hbuf[B_][H_];      // 32 KB (f32 h for final carry)
  __shared__ float  biasl[G_];         // 4 KB
  __shared__ float  bdl[OUT_];
  __shared__ int    lenl[B_];

  const int dir  = blockIdx.x;
  const int tid  = threadIdx.x;
  const int lane = tid & 31;
  const int w    = tid >> 5;           // wave id 0..31

  const float* c0   = dir ? c0b : c0f;
  const float* h0   = dir ? h0b : h0f;
  const float* bias = dir ? bias_b : bias_f;

  // ---- init state -----------------------------------------------------------
  for (int i = tid; i < B_ * H_; i += 1024) {
    const int b = i / H_, j = i % H_;
    const float c = c0[i], h = h0[i];
    cbuf[b][j] = c;
    hbuf[b][j] = h;
    a_buf[b][F_ + j] = (__bf16)h;
  }
  for (int i = tid; i < G_; i += 1024) biasl[i] = bias[i];
  if (tid < OUT_) bdl[tid] = bd[tid];
  if (tid < B_)   lenl[tid] = lens[tid];
  __syncthreads();

  for (int t = 0; t < T_; ++t) {
    // ---- stage x_t (forward: t; backward: (len-1-t) mod T) ------------------
    for (int i = tid; i < B_ * F_; i += 1024) {
      const int b = i >> 6, f = i & 63;
      int tp = t;
      if (dir) { tp = lenl[b] - 1 - t; if (tp < 0) tp += T_; }
      a_buf[b][f] = (__bf16)x[((size_t)b * T_ + tp) * F_ + f];
    }
    __syncthreads();

    // ---- gate GEMM: [32x320] * [320x1024], each wave owns N-tiles 2w,2w+1 ---
    {
      // Loop-variant (laundered) offset from the global kernarg base:
      // streams fragments from L2 each step, no hoist + scratch spill.
      const __bf16* WpT = wsb + launder_off((size_t)dir * W_ELEMS_PER_DIR);
      v8f acc00 = {}, acc01 = {}, acc10 = {}, acc11 = {};
      const int nt0 = 2 * w, nt1 = 2 * w + 1;
#pragma unroll
      for (int kc = 0; kc < KC_; ++kc) {
        const int kbase = kc * 32;
        const v16bf a0 = load_a_frag(&a_buf[(lane & 15) + 0][0],  kbase, lane);
        const v16bf a1 = load_a_frag(&a_buf[(lane & 15) + 16][0], kbase, lane);
        const v16bf b0 = load_b_frag(WpT + ((size_t)(nt0 * KC_ + kc) * 32 + lane) * 16);
        const v16bf b1 = load_b_frag(WpT + ((size_t)(nt1 * KC_ + kc) * 32 + lane) * 16);
        acc00 = wmma_bf16(a0, b0, acc00);
        acc10 = wmma_bf16(a1, b0, acc10);
        acc01 = wmma_bf16(a0, b1, acc01);
        acc11 = wmma_bf16(a1, b1, acc11);
      }
      // scatter accumulators (+bias) into the LDS gate buffer
      const int mrow = (lane & 16) ? 8 : 0;
      const int c0n = nt0 * 16 + (lane & 15);
      const int c1n = nt1 * 16 + (lane & 15);
      const float bi0 = biasl[c0n], bi1 = biasl[c1n];
#pragma unroll
      for (int r = 0; r < 8; ++r) {
        const int m = r + mrow;
        gates[m][c0n]      = acc00[r] + bi0;
        gates[m + 16][c0n] = acc10[r] + bi0;
        gates[m][c1n]      = acc01[r] + bi1;
        gates[m + 16][c1n] = acc11[r] + bi1;
      }
    }
    __syncthreads();

    // ---- pointwise LSTM cell update (gate order: i, f, g, o) ----------------
    {
      const int b  = tid >> 5;
      const int j0 = (tid & 31) * 8;
#pragma unroll
      for (int q = 0; q < 8; ++q) {
        const int j = j0 + q;
        const float ig = gates[b][j];
        const float fg = gates[b][H_ + j];
        const float gg = gates[b][2 * H_ + j];
        const float og = gates[b][3 * H_ + j];
        float c = cbuf[b][j];
        c = sigf(fg) * c + sigf(ig) * tanhfast(gg);
        const float h = sigf(og) * tanhfast(c);
        cbuf[b][j] = c;
        hbuf[b][j] = h;
        a_buf[b][F_ + j] = (__bf16)h;
      }
    }
    __syncthreads();

    // ---- fused output projection: h_t[32x256] * Wd_half[256x64] -------------
    if (w < 8) {
      const __bf16* WdT =
          wsb + launder_off((size_t)WD_BASE_HALF + (size_t)dir * WD_ELEMS_PER_DIR);
      const int mt = w >> 2, nt = w & 3;
      v8f acc = {};
#pragma unroll
      for (int kc = 0; kc < DKC_; ++kc) {
        const v16bf a = load_a_frag(&a_buf[mt * 16 + (lane & 15)][0], F_ + kc * 32, lane);
        const v16bf bw = load_b_frag(WdT + ((size_t)(nt * DKC_ + kc) * 32 + lane) * 16);
        acc = wmma_bf16(a, bw, acc);
      }
      const int o = nt * 16 + (lane & 15);
      const int mrow = (lane & 16) ? 8 : 0;
#pragma unroll
      for (int r = 0; r < 8; ++r) {
        const int b = mt * 16 + r + mrow;
        if (dir == 0) {
          out[OUTS_OFF + ((size_t)b * T_ + t) * OUT_ + o] = acc[r] + bdl[o];
        } else {
          int tp = lenl[b] - 1 - t; if (tp < 0) tp += T_;   // un-flip on scatter
          Pb[((size_t)b * T_ + tp) * OUT_ + o] = acc[r];
        }
      }
    }
    // no barrier needed: next staging touches only the x-region of a_buf
  }

  __syncthreads();
  // ---- final carries: out = [cf | hf | cb | hb | outs] ----------------------
  {
    const size_t base = (size_t)dir * 2 * B_ * H_;
    for (int i = tid; i < B_ * H_; i += 1024) {
      const int b = i / H_, j = i % H_;
      out[base + i]            = cbuf[b][j];
      out[base + B_ * H_ + i]  = hbuf[b][j];
    }
  }
}

// ---- Kernel 3: merge backward partial into the output -----------------------

__global__ void lstm_add_bwd(const float* __restrict__ Pb, float* __restrict__ out) {
  const size_t i = (size_t)blockIdx.x * blockDim.x + threadIdx.x;
  const size_t N = (size_t)B_ * T_ * OUT_;
  if (i < N) out[OUTS_OFF + i] += Pb[i];
}

// ---- launcher ---------------------------------------------------------------

extern "C" void kernel_launch(void* const* d_in, const int* in_sizes, int n_in,
                              void* d_out, int out_size, void* d_ws, size_t ws_size,
                              hipStream_t stream) {
  const float* x    = (const float*)d_in[0];
  const int*   lens = (const int*)d_in[1];
  const float* cf   = (const float*)d_in[2];
  const float* hf   = (const float*)d_in[3];
  const float* cb   = (const float*)d_in[4];
  const float* hb   = (const float*)d_in[5];
  const float* Wi_f = (const float*)d_in[6];
  const float* Wh_f = (const float*)d_in[7];
  const float* b_f  = (const float*)d_in[8];
  const float* Wi_b = (const float*)d_in[9];
  const float* Wh_b = (const float*)d_in[10];
  const float* b_b  = (const float*)d_in[11];
  const float* Wd   = (const float*)d_in[12];
  const float* bd   = (const float*)d_in[13];

  float*  out = (float*)d_out;
  __bf16* wsb = (__bf16*)d_ws;
  float*  Pb  = (float*)((char*)d_ws + PB_BYTE_OFF);

  const int pack_elems = 2 * W_ELEMS_PER_DIR + 2 * WD_ELEMS_PER_DIR;  // 688128
  lstm_prepack<<<(pack_elems + 255) / 256, 256, 0, stream>>>(
      Wi_f, Wh_f, Wi_b, Wh_b, Wd, wsb);

  bilstm_recurrent<<<2, 1024, 0, stream>>>(
      x, lens, cf, hf, cb, hb, b_f, b_b, bd, wsb, Pb, out);

  const size_t n = (size_t)B_ * T_ * OUT_;
  lstm_add_bwd<<<(unsigned)((n + 255) / 256), 256, 0, stream>>>(Pb, out);
}
